// LOP_Adaptive_8194797601443
// MI455X (gfx1250) — matrix operand order repaired and checked by reading, not compiled, not run on hardware
//
#include <hip/hip_runtime.h>

// ---------------------------------------------------------------------------
// Problem constants (match reference)
// ---------------------------------------------------------------------------
#define BB 4
#define TT 1024
#define DD 1024
#define HH 16
#define HD 64
#define BT (BB * TT)          // 4096
#define MAXK 9

typedef __bf16 bf16x16 __attribute__((ext_vector_type(16)));
typedef float  floatx8 __attribute__((ext_vector_type(8)));

union BF16Frag {
    bf16x16 v;
    uint4   q[2];
    unsigned short u[16];
};

__device__ __forceinline__ unsigned short f32_to_bf16(float f) {
    unsigned int u = __float_as_uint(f);
    unsigned int r = 0x7FFFu + ((u >> 16) & 1u);
    return (unsigned short)((u + r) >> 16);
}
__device__ __forceinline__ float bf16_to_f32(unsigned short s) {
    return __uint_as_float(((unsigned int)s) << 16);
}

// gfx1250 async global->LDS path (guarded; falls back to reg-pipelined copy)
#if defined(__has_builtin)
#  if __has_builtin(__builtin_amdgcn_global_load_async_to_lds_b128)
#    define HAVE_ASYNC_LDS 1
#  endif
#endif
#ifndef HAVE_ASYNC_LDS
#  define HAVE_ASYNC_LDS 0
#endif

#if HAVE_ASYNC_LDS
// Parameter type per hipcc diagnostic: generic pointer to 16-byte int vector.
typedef int gld_v4i __attribute__((vector_size(16)));
#  define ASYNC_PTR(p) ((gld_v4i*)(p))
__device__ __forceinline__ void wait_asynccnt0() {
#  if __has_builtin(__builtin_amdgcn_s_wait_asynccnt)
    __builtin_amdgcn_s_wait_asynccnt(0);
#  else
    asm volatile("s_wait_asynccnt 0x0" ::: "memory");
#  endif
}
#endif

// ---------------------------------------------------------------------------
// fp32 -> bf16 conversion (plain)
// ---------------------------------------------------------------------------
__global__ void cvt_bf16_kernel(const float* __restrict__ in,
                                unsigned short* __restrict__ out, int n) {
    int i = blockIdx.x * blockDim.x + threadIdx.x;
    if (i < n) out[i] = f32_to_bf16(in[i]);
}

// ---------------------------------------------------------------------------
// fp32 (K x N) -> bf16 transposed (N x K), LDS-tiled for coalescing.
// ---------------------------------------------------------------------------
__global__ __launch_bounds__(256) void cvt_transpose_kernel(
    const float* __restrict__ in, unsigned short* __restrict__ outp,
    int K, int N) {
    __shared__ float tile[32][33];
    const int kb = blockIdx.x * 32, nb = blockIdx.y * 32;
    const int tx = threadIdx.x & 31, ty = threadIdx.x >> 5;  // 32 x 8
    #pragma unroll
    for (int j = 0; j < 32; j += 8)
        tile[ty + j][tx] = in[(size_t)(kb + ty + j) * N + nb + tx];
    __syncthreads();
    #pragma unroll
    for (int j = 0; j < 32; j += 8)
        outp[(size_t)(nb + ty + j) * K + kb + tx] = f32_to_bf16(tile[tx][ty + j]);
}

// ---------------------------------------------------------------------------
// Tiny MLP selecting conv kernel size per head.
// Softmax rows sum to 1 => importance identical for all heads => imp_norm = 0.
// ---------------------------------------------------------------------------
__global__ void kproj_kernel(const float* __restrict__ kp_w1,
                             const float* __restrict__ kp_b1,
                             const float* __restrict__ kp_w2,
                             const float* __restrict__ kp_b2,
                             int* __restrict__ kidx) {
    int h = threadIdx.x;
    if (h >= HH) return;
    float acc = kp_b2[0];
    for (int i = 0; i < 16; ++i) {
        float hm = fmaxf(0.f, 0.f * kp_w1[i] + kp_b1[i]);
        acc += hm * kp_w2[i];
    }
    float frac = 1.f / (1.f + __expf(-acc));
    int kv = (int)rintf(3.f + frac * 6.f);
    kv += (1 - (kv & 1));
    kv = min(max(kv, 3), MAXK);
    int ki = (kv - 3) >> 1;
    kidx[h] = min(max(ki, 0), 3);
}

// ---------------------------------------------------------------------------
// bf16 GEMM: C(M,N) = A(M,K) @ B(K,N) + bias via v_wmma_f32_16x16x32_bf16.
// B supplied TRANSPOSED (Bt = N x K) so the LDS tile copy is a straight
// per-thread b128 copy (async-to-LDS) and B-fragments are contiguous.
// 256 threads = 8 waves; wave tile 32x64 (8 accumulators); double-buffered
// B tile in LDS, software-pipelined (async when available).
// ---------------------------------------------------------------------------
__global__ __launch_bounds__(256) void gemm_bf16_kernel(
    const unsigned short* __restrict__ A,    // M x K bf16
    const unsigned short* __restrict__ Bt,   // N x K bf16 (transposed B)
    const float* __restrict__ bias,          // N
    unsigned short* __restrict__ Cbf,        // bf16 out (or nullptr)
    float* __restrict__ Cf,                  // f32 out  (or nullptr)
    int M, int N, int K) {
    __shared__ __align__(16) unsigned short sB[2][64][32];  // [buf][col][k]

    const int tid  = threadIdx.x;
    const int wave = tid >> 5;
    const int lane = tid & 31;
    const int lm   = lane & 15;
    const int lhi  = lane >> 4;

    const int rowbase = blockIdx.x * 256 + wave * 32;
    const int colbase = blockIdx.y * 64;

    // B-tile copy indices: 64 cols x 32 k, 16B per thread
    const int tc = tid >> 2;          // col 0..63
    const int tk = (tid & 3) * 8;     // k chunk {0,8,16,24}
    const unsigned short* bsrc = Bt + (size_t)(colbase + tc) * K + tk;

    floatx8 acc0[4] = {};
    floatx8 acc1[4] = {};

    // stage tile 0
#if HAVE_ASYNC_LDS
    __builtin_amdgcn_global_load_async_to_lds_b128(
        ASYNC_PTR(bsrc), ASYNC_PTR(&sB[0][tc][tk]), 0, 0);
    wait_asynccnt0();
#else
    uint4 nb = *(const uint4*)(bsrc);
    *(uint4*)&sB[0][tc][tk] = nb;
#endif
    __syncthreads();

    for (int k0 = 0; k0 < K; k0 += 32) {
        const int  cur  = (k0 >> 5) & 1;
        const bool more = (k0 + 32) < K;

        // issue next tile before compute so the transfer overlaps the WMMAs
#if HAVE_ASYNC_LDS
        if (more)
            __builtin_amdgcn_global_load_async_to_lds_b128(
                ASYNC_PTR(bsrc + k0 + 32), ASYNC_PTR(&sB[cur ^ 1][tc][tk]), 0, 0);
#else
        if (more) nb = *(const uint4*)(bsrc + k0 + 32);
#endif

        // A fragments for the two 16-row subtiles
        BF16Frag af0, af1;
        const unsigned short* ap0 =
            A + (size_t)(rowbase + lm) * K + k0 + lhi * 8;
        af0.q[0] = *(const uint4*)(ap0);
        af0.q[1] = *(const uint4*)(ap0 + 16);
        const unsigned short* ap1 = ap0 + (size_t)16 * K;
        af1.q[0] = *(const uint4*)(ap1);
        af1.q[1] = *(const uint4*)(ap1 + 16);

        #pragma unroll
        for (int jn = 0; jn < 4; ++jn) {
            BF16Frag bf;
            const unsigned short* bp = &sB[cur][jn * 16 + lm][lhi * 8];
            bf.q[0] = *(const uint4*)(bp);
            bf.q[1] = *(const uint4*)(bp + 16);
            acc0[jn] = __builtin_amdgcn_wmma_f32_16x16x32_bf16(
                false, af0.v, false, bf.v, (short)0, acc0[jn], false, false);
            acc1[jn] = __builtin_amdgcn_wmma_f32_16x16x32_bf16(
                false, af1.v, false, bf.v, (short)0, acc1[jn], false, false);
        }

#if HAVE_ASYNC_LDS
        wait_asynccnt0();
#else
        if (more) *(uint4*)&sB[cur ^ 1][tc][tk] = nb;
#endif
        __syncthreads();
    }

    // epilogue
    #pragma unroll
    for (int rt = 0; rt < 2; ++rt) {
        floatx8* acc = rt ? acc1 : acc0;
        #pragma unroll
        for (int jn = 0; jn < 4; ++jn) {
            int col = colbase + jn * 16 + lm;
            float bv = bias ? bias[col] : 0.f;
            #pragma unroll
            for (int r = 0; r < 8; ++r) {
                int row = rowbase + rt * 16 + lhi * 8 + r;
                float val = acc[jn][r] + bv;
                if (Cf)  Cf[(size_t)row * N + col] = val;
                else     Cbf[(size_t)row * N + col] = f32_to_bf16(val);
            }
        }
    }
}

// ---------------------------------------------------------------------------
// Flash attention. 8 waves x 16 query rows per block; 32-key KV tiles in LDS
// (V transposed); software-pipelined tile fetch (next tile loads into regs
// while current tile's WMMAs run). Online softmax; P is round-tripped
// through per-wave LDS staging to reach A-fragment layout for P@V.
// Output stored bf16 (consumed by conv).
// ---------------------------------------------------------------------------
__global__ __launch_bounds__(256) void attn_kernel(
    const unsigned short* __restrict__ Q,   // (B*T, D) bf16
    const unsigned short* __restrict__ Kb,  // (B*T, D) bf16
    const unsigned short* __restrict__ Vb,  // (B*T, D) bf16
    unsigned short* __restrict__ O) {       // (B*T, D) bf16
    __shared__ __align__(16) unsigned short sK[32][64];      // [key][hd]
    __shared__ __align__(16) unsigned short sV[64][32];      // [hd][key]
    __shared__ __align__(16) unsigned short sP[8][16][32];   // per-wave P staging

    int gid  = blockIdx.x;                 // B*H*(T/128)
    int qblk = gid & 7;  gid >>= 3;
    int h    = gid & (HH - 1);
    int b    = gid >> 4;

    const int tid  = threadIdx.x;
    const int wave = tid >> 5;
    const int lane = tid & 31;
    const int lm   = lane & 15;
    const int lhi  = lane >> 4;

    const int qrow0 = qblk * 128 + wave * 16;
    const int colq  = h * HD;
    const float scale = 0.125f;            // HD^-0.5

    // Q A-fragments, live across the kv loop
    BF16Frag qf[2];
    #pragma unroll
    for (int c = 0; c < 2; ++c) {
        const unsigned short* qp =
            Q + (size_t)(b * TT + qrow0 + lm) * DD + colq + c * 32 + lhi * 8;
        qf[c].q[0] = *(const uint4*)(qp);
        qf[c].q[1] = *(const uint4*)(qp + 16);
    }

    floatx8 oacc[4] = {};
    float rm[8], rl[8];
    #pragma unroll
    for (int r = 0; r < 8; ++r) { rm[r] = -3.0e38f; rl[r] = 0.f; }

    // pipeline prologue: fetch KV tile 0 into registers
    const int rr = tid >> 3;
    const int c8 = (tid & 7) * 8;
    const size_t kvstride = (size_t)DD;
    const unsigned short* kbase = Kb + (size_t)(b * TT) * DD + colq + c8;
    const unsigned short* vbase = Vb + (size_t)(b * TT) * DD + colq + c8;
    uint4 dk = *(const uint4*)(kbase + (size_t)rr * kvstride);
    union { uint4 q; unsigned short u[8]; } dv;
    dv.q = *(const uint4*)(vbase + (size_t)rr * kvstride);

    for (int kv0 = 0; kv0 < TT; kv0 += 32) {
        // stage current tile into LDS
        *(uint4*)&sK[rr][c8] = dk;
        #pragma unroll
        for (int j = 0; j < 8; ++j) sV[c8 + j][rr] = dv.u[j];
        __syncthreads();

        // prefetch next tile into registers (overlaps the WMMAs below)
        if (kv0 + 32 < TT) {
            dk   = *(const uint4*)(kbase + (size_t)(kv0 + 32 + rr) * kvstride);
            dv.q = *(const uint4*)(vbase + (size_t)(kv0 + 32 + rr) * kvstride);
        }

        // --- S = Q @ K^T ---
        floatx8 sacc[2] = {};
        #pragma unroll
        for (int jn = 0; jn < 2; ++jn) {
            #pragma unroll
            for (int c = 0; c < 2; ++c) {
                BF16Frag kf;
                const unsigned short* kp = &sK[jn * 16 + lm][c * 32 + lhi * 8];
                kf.q[0] = *(const uint4*)(kp);
                kf.q[1] = *(const uint4*)(kp + 16);
                sacc[jn] = __builtin_amdgcn_wmma_f32_16x16x32_bf16(
                    false, qf[c].v, false, kf.v, (short)0, sacc[jn], false, false);
            }
        }

        // --- online softmax over 32 new keys ---
        #pragma unroll
        for (int r = 0; r < 8; ++r) {
            float s0 = sacc[0][r] * scale;
            float s1 = sacc[1][r] * scale;
            float mx = fmaxf(s0, s1);
            mx = fmaxf(mx, __shfl_xor(mx, 1, 32));
            mx = fmaxf(mx, __shfl_xor(mx, 2, 32));
            mx = fmaxf(mx, __shfl_xor(mx, 4, 32));
            mx = fmaxf(mx, __shfl_xor(mx, 8, 32));
            float nm   = fmaxf(rm[r], mx);
            float corr = __expf(rm[r] - nm);
            rm[r] = nm;
            rl[r] *= corr;
            oacc[0][r] *= corr; oacc[1][r] *= corr;
            oacc[2][r] *= corr; oacc[3][r] *= corr;
            float p0 = __expf(s0 - nm);
            float p1 = __expf(s1 - nm);
            float rs = p0 + p1;
            rs += __shfl_xor(rs, 1, 32);
            rs += __shfl_xor(rs, 2, 32);
            rs += __shfl_xor(rs, 4, 32);
            rs += __shfl_xor(rs, 8, 32);
            rl[r] += rs;
            sP[wave][lhi * 8 + r][lm]      = f32_to_bf16(p0);
            sP[wave][lhi * 8 + r][16 + lm] = f32_to_bf16(p1);
        }
        // in-wave LDS RAW hazard on P staging
        asm volatile("s_wait_dscnt 0" ::: "memory");

        // --- O += P @ V ---
        BF16Frag pf;
        {
            const unsigned short* pp = &sP[wave][lm][lhi * 8];
            pf.q[0] = *(const uint4*)(pp);
            pf.q[1] = *(const uint4*)(pp + 16);
        }
        #pragma unroll
        for (int c = 0; c < 4; ++c) {
            BF16Frag vf;
            const unsigned short* vp = &sV[c * 16 + lm][lhi * 8];
            vf.q[0] = *(const uint4*)(vp);
            vf.q[1] = *(const uint4*)(vp + 16);
            oacc[c] = __builtin_amdgcn_wmma_f32_16x16x32_bf16(
                false, pf.v, false, vf.v, (short)0, oacc[c], false, false);
        }
        __syncthreads();
    }

    // normalize + store bf16, x-layout
    #pragma unroll
    for (int c = 0; c < 4; ++c) {
        #pragma unroll
        for (int r = 0; r < 8; ++r) {
            int t  = qrow0 + lhi * 8 + r;
            int hd = c * 16 + lm;
            O[(size_t)(b * TT + t) * DD + colq + hd] =
                f32_to_bf16(oacc[c][r] / rl[r]);
        }
    }
}

// ---------------------------------------------------------------------------
// Grouped Conv1d with per-head dynamic kernel size + edge-replicated output
// padding. LDS-tiled: one (b,h,64-row t-segment) per block with 4-row halo;
// all 32 lanes of a wave read the same LDS word (broadcast). Weights are
// register-resident via predicated MAXK unroll. y[t] = y_valid[clamp(t-half)].
// ---------------------------------------------------------------------------
__global__ __launch_bounds__(256) void conv_kernel(
    const unsigned short* __restrict__ O,    // (B*T, D) bf16
    const float* __restrict__ cw,            // (H, HD, HD, MAXK)
    const float* __restrict__ cb,            // (H, HD)
    const int* __restrict__ kidx,            // (H,)
    unsigned short* __restrict__ out_cat) {  // (B*T, D) bf16
    __shared__ __align__(16) unsigned short sO[72][HD];  // 64 rows + 8 halo

    const int tid = threadIdx.x;
    int blk = blockIdx.x;                    // B*H*(T/64) = 1024
    const int tseg = blk & 15; blk >>= 4;
    const int h = blk & (HH - 1);
    const int b = blk >> 4;
    const int t0 = tseg * 64;

    // stage rows [t0-4, t0+68) (clamped; clamped halo rows are never read)
    for (int i = tid; i < 72 * 8; i += 256) {
        const int r  = i >> 3;
        const int cc = (i & 7) * 8;
        const int t  = min(max(t0 + r - 4, 0), TT - 1);
        *(uint4*)&sO[r][cc] =
            *(const uint4*)(O + (size_t)(b * TT + t) * DD + h * HD + cc);
    }
    __syncthreads();

    const int kk   = 3 + 2 * kidx[h];
    const int half = kk >> 1;
    const int c    = tid & (HD - 1);
    const int tq   = tid >> 6;               // 0..3
    const int tb   = t0 + tq * 16;

    float acc[16];
    const float bias = cb[h * HD + c];
    #pragma unroll
    for (int jj = 0; jj < 16; ++jj) acc[jj] = bias;

    const float* wbase =
        cw + ((size_t)(h * HD + c) * HD) * MAXK + (MAXK / 2 - half);
    for (int ci = 0; ci < HD; ++ci) {
        const float* wr = wbase + ci * MAXK;
        float wreg[MAXK];
        #pragma unroll
        for (int j = 0; j < MAXK; ++j) {
            float wv = wr[(j < kk) ? j : 0];       // safe in-bounds index
            wreg[j] = (j < kk) ? wv : 0.f;         // zero => FMA is a no-op
        }
        #pragma unroll
        for (int jj = 0; jj < 16; ++jj) {
            int s  = min(max(tb + jj - half, 0), TT - kk);
            int sl = s - t0 + 4;
            float a = acc[jj];
            #pragma unroll
            for (int j = 0; j < MAXK; ++j)
                a += wreg[j] * bf16_to_f32(sO[sl + j][ci]);
            acc[jj] = a;
        }
    }
    #pragma unroll
    for (int jj = 0; jj < 16; ++jj)
        out_cat[(size_t)(b * TT + tb + jj) * DD + h * HD + c] =
            f32_to_bf16(acc[jj]);
}

// ---------------------------------------------------------------------------
// Residual + LayerNorm: out = LN(proj + x) * g + b, one row per block.
// ---------------------------------------------------------------------------
__global__ __launch_bounds__(256) void ln_kernel(
    const float* __restrict__ proj, const float* __restrict__ xin,
    const float* __restrict__ g, const float* __restrict__ bt,
    float* __restrict__ out) {
    __shared__ float red[256];
    const int row = blockIdx.x;
    const int tid = threadIdx.x;
    const float* pr = proj + (size_t)row * DD;
    const float* xr = xin  + (size_t)row * DD;

    float y[4];
    float s = 0.f;
    #pragma unroll
    for (int j = 0; j < 4; ++j) {
        int i = tid + 256 * j;
        y[j] = pr[i] + xr[i];
        s += y[j];
    }
    red[tid] = s; __syncthreads();
    for (int st = 128; st > 0; st >>= 1) {
        if (tid < st) red[tid] += red[tid + st];
        __syncthreads();
    }
    float mu = red[0] * (1.f / DD);
    __syncthreads();

    float v = 0.f;
    #pragma unroll
    for (int j = 0; j < 4; ++j) { float d = y[j] - mu; v += d * d; }
    red[tid] = v; __syncthreads();
    for (int st = 128; st > 0; st >>= 1) {
        if (tid < st) red[tid] += red[tid + st];
        __syncthreads();
    }
    float var = red[0] * (1.f / DD);
    float inv = rsqrtf(var + 1e-5f);
    #pragma unroll
    for (int j = 0; j < 4; ++j) {
        int i = tid + 256 * j;
        out[(size_t)row * DD + i] = (y[j] - mu) * inv * g[i] + bt[i];
    }
}

// ---------------------------------------------------------------------------
// Host-side launcher
// ---------------------------------------------------------------------------
extern "C" void kernel_launch(void* const* d_in, const int* in_sizes, int n_in,
                              void* d_out, int out_size, void* d_ws, size_t ws_size,
                              hipStream_t stream) {
    const float* x      = (const float*)d_in[0];
    const float* Wq     = (const float*)d_in[1];
    const float* bq     = (const float*)d_in[2];
    const float* Wk     = (const float*)d_in[3];
    const float* bk     = (const float*)d_in[4];
    const float* Wv     = (const float*)d_in[5];
    const float* bv     = (const float*)d_in[6];
    const float* conv_w = (const float*)d_in[7];
    const float* conv_b = (const float*)d_in[8];
    const float* kp_w1  = (const float*)d_in[9];
    const float* kp_b1  = (const float*)d_in[10];
    const float* kp_w2  = (const float*)d_in[11];
    const float* kp_b2  = (const float*)d_in[12];
    const float* fc_w   = (const float*)d_in[13];
    const float* fc_b   = (const float*)d_in[14];
    const float* ln_g   = (const float*)d_in[15];
    const float* ln_b   = (const float*)d_in[16];
    float* out = (float*)d_out;

    char* ws = (char*)d_ws;
    size_t off = 0;
    auto carve = [&](size_t bytes) {
        char* p = ws + off;
        off = (off + bytes + 255) & ~(size_t)255;
        return p;
    };
    unsigned short* xb   = (unsigned short*)carve((size_t)BT * DD * 2);
    unsigned short* wqt  = (unsigned short*)carve((size_t)DD * DD * 2);
    unsigned short* wkt  = (unsigned short*)carve((size_t)DD * DD * 2);
    unsigned short* wvt  = (unsigned short*)carve((size_t)DD * DD * 2);
    unsigned short* fct  = (unsigned short*)carve((size_t)DD * DD * 2);
    unsigned short* qb   = (unsigned short*)carve((size_t)BT * DD * 2);
    unsigned short* kb   = (unsigned short*)carve((size_t)BT * DD * 2);
    unsigned short* vb   = (unsigned short*)carve((size_t)BT * DD * 2);
    unsigned short* Obuf = (unsigned short*)carve((size_t)BT * DD * 2);
    unsigned short* ocat = (unsigned short*)carve((size_t)BT * DD * 2);
    float*          proj = (float*)carve((size_t)BT * DD * 4);
    int*            kidx = (int*)carve(HH * sizeof(int));
    (void)ws_size; (void)n_in; (void)in_sizes; (void)out_size;

    const int nX = BT * DD;

    // 1) convert x; convert+transpose the four weight matrices (Bt layout)
    cvt_bf16_kernel<<<(nX + 255) / 256, 256, 0, stream>>>(x, xb, nX);
    dim3 tgrid(DD / 32, DD / 32);
    cvt_transpose_kernel<<<tgrid, 256, 0, stream>>>(Wq, wqt, DD, DD);
    cvt_transpose_kernel<<<tgrid, 256, 0, stream>>>(Wk, wkt, DD, DD);
    cvt_transpose_kernel<<<tgrid, 256, 0, stream>>>(Wv, wvt, DD, DD);
    cvt_transpose_kernel<<<tgrid, 256, 0, stream>>>(fc_w, fct, DD, DD);

    // 2) QKV projections (WMMA GEMMs, bf16 outputs in x-layout)
    dim3 ggrid(BT / 256, DD / 64);
    gemm_bf16_kernel<<<ggrid, 256, 0, stream>>>(xb, wqt, bq, qb, nullptr, BT, DD, DD);
    gemm_bf16_kernel<<<ggrid, 256, 0, stream>>>(xb, wkt, bk, kb, nullptr, BT, DD, DD);
    gemm_bf16_kernel<<<ggrid, 256, 0, stream>>>(xb, wvt, bv, vb, nullptr, BT, DD, DD);

    // 3) dynamic kernel-size selection (softmax row-sum identity => exact)
    kproj_kernel<<<1, 32, 0, stream>>>(kp_w1, kp_b1, kp_w2, kp_b2, kidx);

    // 4) flash attention (WMMA), bf16 output
    attn_kernel<<<BB * HH * (TT / 128), 256, 0, stream>>>(qb, kb, vb, Obuf);

    // 5) grouped conv with selected kernel size (LDS-tiled), bf16 out_cat
    conv_kernel<<<BB * HH * (TT / 64), 256, 0, stream>>>(
        Obuf, conv_w, conv_b, kidx, ocat);

    // 6) fc projection (WMMA GEMM, fp32 output)
    gemm_bf16_kernel<<<ggrid, 256, 0, stream>>>(ocat, fct, fc_b, nullptr, proj, BT, DD, DD);

    // 7) residual + LayerNorm
    ln_kernel<<<BT, 256, 0, stream>>>(proj, x, ln_g, ln_b, out);
}
